// MixerModel_add_32882269618430
// MI455X (gfx1250) — matrix operand order repaired and checked by reading, not compiled, not run on hardware
//
#include <hip/hip_runtime.h>
#include <hip/hip_bf16.h>
#include <stdint.h>

#define B_      8
#define LSEQ    4096
#define DMODEL  384
#define DINNER  768
#define DSTATE  16
#define MROWS   (B_*LSEQ)        /* 32768 rows */
#define NXZ     1536
#define NXDB    64               /* 56 padded to 64 */
#define KDT     32               /* 24 padded to 32 */

typedef __attribute__((ext_vector_type(16))) _Float16 v16h;
typedef __attribute__((ext_vector_type(8)))  float    v8f;
typedef __attribute__((ext_vector_type(4)))  unsigned v4u;
typedef __attribute__((ext_vector_type(8)))  int      v8i;
typedef __attribute__((ext_vector_type(4)))  int      v4i;

__device__ __forceinline__ float siluf(float x) { return x / (1.f + __expf(-x)); }

// ---------------------------------------------------------------- init: h = ids + pos
__global__ void k_init(const float* __restrict__ a, const float* __restrict__ b,
                       float* __restrict__ h, int n) {
  int stride = gridDim.x * blockDim.x;
  for (int i = blockIdx.x * blockDim.x + threadIdx.x; i < n; i += stride)
    h[i] = a[i] + b[i];
}

// ------------------------------------------------- weight convert (with K/N zero padding)
__global__ void k_cvtw(const float* __restrict__ Win, const float* __restrict__ Wx,
                       const float* __restrict__ Wdt, const float* __restrict__ Wout,
                       _Float16* __restrict__ Win16, _Float16* __restrict__ Wx16,
                       _Float16* __restrict__ Wdt16, _Float16* __restrict__ Wout16) {
  const int n0 = 4*1536*384, n1 = 4*64*768, n2 = 4*768*32, n3 = 4*384*768;
  const int tot = n0 + n1 + n2 + n3;
  int stride = gridDim.x * blockDim.x;
  for (int i = blockIdx.x * blockDim.x + threadIdx.x; i < tot; i += stride) {
    if (i < n0) {
      Win16[i] = (_Float16)Win[i];
    } else if (i < n0 + n1) {
      int j = i - n0; int l = j / (64*768); int e = (j / 768) % 64; int d = j % 768;
      Wx16[j] = (_Float16)(e < 56 ? Wx[(l*56 + e)*768 + d] : 0.f);
    } else if (i < n0 + n1 + n2) {
      int j = i - n0 - n1; int l = j / (768*32); int dd = (j / 32) % 768; int r = j % 32;
      Wdt16[j] = (_Float16)(r < 24 ? Wdt[(l*768 + dd)*24 + r] : 0.f);
    } else {
      int j = i - n0 - n1 - n2;
      Wout16[j] = (_Float16)Wout[j];
    }
  }
}

// ---------------------------------------------------------------- rank-sort eigenvectors
__global__ __launch_bounds__(512) void k_sort(const float* __restrict__ eig,
                                              int* __restrict__ perm, int* __restrict__ inv) {
  __shared__ float se[512];
  const int bk = blockIdx.x;            // b*4 + k
  const int b = bk >> 2, k = bk & 3;
  const int n = threadIdx.x;
  float v = eig[((size_t)b*512 + n)*4 + k];
  se[n] = v;
  __syncthreads();
  int r = 0;
  for (int m = 0; m < 512; ++m) {
    float u = se[m];
    r += (u < v) || (u == v && m < n);  // stable rank
  }
  inv[bk*512 + n] = r;
  perm[bk*512 + r] = n;
}

// ------------------------------------------------- build 8-way gather table per output row
__global__ void k_gidx(const int* __restrict__ perm, const int* __restrict__ inv,
                       const int* __restrict__ revp, int* __restrict__ g) {
  int t = blockIdx.x * blockDim.x + threadIdx.x;
  if (t >= B_ * LSEQ) return;
  int b = t >> 12, l = t & 4095;
  int rev = revp[0];
  int l2 = (rev && l >= 2048) ? (4095 - l) : l;
  int j = l2 >> 9, m = l2 & 511;
  int p = perm[(b*4 + j)*512 + m];
#pragma unroll
  for (int k = 0; k < 4; ++k) {
    int r = inv[(b*4 + k)*512 + p];
    g[t*8 + 2*k]     = k*512 + r;               // forward stream
    g[t*8 + 2*k + 1] = (k + 4)*512 + (511 - r); // reversed stream
  }
}

// ---------------------------------------------------------------- residual + LayerNorm
__global__ __launch_bounds__(256) void k_ln(const float* __restrict__ hin,
    float* __restrict__ res, _Float16* __restrict__ hn,
    const float* __restrict__ w, const float* __restrict__ bb, int first) {
  int row = (blockIdx.x << 3) + (threadIdx.x >> 5);
  int lane = threadIdx.x & 31;
  size_t base = (size_t)row * DMODEL;
  float v[12]; float s = 0.f, sq = 0.f;
#pragma unroll
  for (int q = 0; q < 12; ++q) {
    int d = q*32 + lane;
    float x = hin[base + d];
    if (!first) x += res[base + d];
    res[base + d] = x;
    v[q] = x; s += x; sq += x * x;
  }
#pragma unroll
  for (int o = 16; o > 0; o >>= 1) { s += __shfl_xor(s, o, 32); sq += __shfl_xor(sq, o, 32); }
  float mean = s * (1.f/384.f);
  float var  = sq * (1.f/384.f) - mean*mean;
  float rs   = rsqrtf(var + 1e-5f);
#pragma unroll
  for (int q = 0; q < 12; ++q) {
    int d = q*32 + lane;
    hn[base + d] = (_Float16)((v[q] - mean) * rs * w[d] + bb[d]);
  }
}

__global__ __launch_bounds__(256) void k_ln_final(const float* __restrict__ hin,
    const float* __restrict__ res, float* __restrict__ out,
    const float* __restrict__ w, const float* __restrict__ bb) {
  int row = (blockIdx.x << 3) + (threadIdx.x >> 5);
  int lane = threadIdx.x & 31;
  size_t base = (size_t)row * DMODEL;
  float v[12]; float s = 0.f, sq = 0.f;
#pragma unroll
  for (int q = 0; q < 12; ++q) {
    int d = q*32 + lane;
    float x = hin[base + d] + res[base + d];
    v[q] = x; s += x; sq += x * x;
  }
#pragma unroll
  for (int o = 16; o > 0; o >>= 1) { s += __shfl_xor(s, o, 32); sq += __shfl_xor(sq, o, 32); }
  float mean = s * (1.f/384.f);
  float var  = sq * (1.f/384.f) - mean*mean;
  float rs   = rsqrtf(var + 1e-5f);
#pragma unroll
  for (int q = 0; q < 12; ++q) {
    int d = q*32 + lane;
    out[base + d] = (v[q] - mean) * rs * w[d] + bb[d];
  }
}

// ---------------------------------------------------------------- WMMA GEMM
// C[M,N] = A[M,K] (f16, row-major) * Bw[N,K]^T (f16, row-major N x K), fp32 accum.
// Block = 256 thr = 8 waves; each wave does one 16x16 tile; block tile = 128 x 16.
// The 16 x K weight strip for this block's N-tile is contiguous in Bw -> staged into
// LDS with one Tensor-Data-Mover descriptor (D# per ISA §8), fallback = coop copy.
// EPI: 0 = plain store, 1 = softplus(x + bias[n]).
template<int EPI>
__global__ __launch_bounds__(256) void k_gemm(const _Float16* __restrict__ A,
    const _Float16* __restrict__ Bw, float* __restrict__ C,
    int M, int N, int K, const float* __restrict__ bias) {
  __shared__ _Float16 sB[16 * 768];            // max strip: 16 x 768 f16 = 24 KB
  const int tileN = blockIdx.y << 4;
  const int strip = 16 * K;                    // elements, contiguous in Bw
#if __has_builtin(__builtin_amdgcn_tensor_load_to_lds) && __has_builtin(__builtin_amdgcn_s_wait_tensorcnt)
  if (threadIdx.x < 32) {
    uint64_t ga  = (uint64_t)(uintptr_t)(const void*)(Bw + (size_t)tileN * K);
    uint32_t lds = (uint32_t)(size_t)(void*)sB;  // low 32 bits of shared addr = LDS byte addr
    // D# group0: count=1 | lds_addr | global_addr[56:0] | type=2
    v4u g0; g0[0] = 1u; g0[1] = lds;
    g0[2] = (uint32_t)ga;
    g0[3] = ((uint32_t)(ga >> 32) & 0x1FFFFFFu) | (2u << 30);
    // D# group1: data_size=1 (2B), 1-D tile: tensor_dim0 = tile_dim0 = strip, dim1 = 1
    v8i g1;
    g1[0] = (int)(1u << 16);                               // [17:16] data_size = 1 (2 bytes)
    g1[1] = (int)(((uint32_t)strip & 0xFFFFu) << 16);      // tensor_dim0[15:0] @ bits 63:48
    g1[2] = (int)((((uint32_t)strip >> 16) & 0xFFFFu) | (1u << 16)); // dim0 hi, tensor_dim1=1
    g1[3] = (int)(((uint32_t)strip & 0xFFFFu) << 16);      // tile_dim0 @ bits 127:112
    g1[4] = 0;                                             // tile_dim1/2 unused
    g1[5] = strip;                                         // tensor_dim0_stride lo32
    g1[6] = 0; g1[7] = 0;
    v4i g2 = {0, 0, 0, 0}; v4i g3 = {0, 0, 0, 0};
    v8i g4 = {0, 0, 0, 0, 0, 0, 0, 0};                     // 6-arg form: extra group
    __builtin_amdgcn_tensor_load_to_lds(g0, g1, g2, g3, g4, 0);
    __builtin_amdgcn_s_wait_tensorcnt(0);
  }
#else
  for (int i = threadIdx.x; i < strip; i += 256) sB[i] = Bw[(size_t)tileN * K + i];
#endif
  __syncthreads();

  const int lane = threadIdx.x & 31;
  const int half = lane >> 4;
  const int mrow = (blockIdx.x << 7) + ((threadIdx.x >> 5) << 4) + (lane & 15);
  const int ncol = lane & 15;
  const _Float16* Ap = A + (size_t)mrow * K;
  const _Float16* Bp = sB + ncol * K;
  v8f acc = {};
  for (int kb = 0; kb < K; kb += 32) {
    v16h a, b;
#pragma unroll
    for (int i = 0; i < 8; ++i) {
      // A 16x32 f16 layout: VGPR i holds K = (i>>2)*16 + half*8 + (i&3)*2, +1
      int ka = kb + ((i >> 2) << 4) + (half << 3) + ((i & 3) << 1);
      a[2*i]     = Ap[ka];
      a[2*i + 1] = Ap[ka + 1];
      // B 32x16 f16 layout: VGPR i holds K = (i>>2)*16 + half*8 + (i&3)*2, +1 ; N = lane&15
      int kbb = kb + ((i >> 2) << 4) + (half << 3) + ((i & 3) << 1);
      b[2*i]     = Bp[kbb];
      b[2*i + 1] = Bp[kbb + 1];
    }
    acc = __builtin_amdgcn_wmma_f32_16x16x32_f16(false, a, false, b, (short)0, acc,
                                                 false, false);
  }
  const int nout = tileN + ncol;
#pragma unroll
  for (int v = 0; v < 8; ++v) {
    int row = (blockIdx.x << 7) + ((threadIdx.x >> 5) << 4) + v + (half << 3);
    float x = acc[v];
    if (EPI == 1) {               // softplus(x + b_dt)
      x += bias[nout];
      x = (x > 20.f) ? x : __logf(1.f + __expf(x));
    }
    C[(size_t)row * N + nout] = x;
  }
}

// ---------------------------------------------------------------- causal conv + SiLU
__global__ void k_conv(const float* __restrict__ xz, const float* __restrict__ cw,
                       const float* __restrict__ cb, _Float16* __restrict__ xc, int total) {
  int stride = gridDim.x * blockDim.x;
  for (int i = blockIdx.x * blockDim.x + threadIdx.x; i < total; i += stride) {
    int c  = i % DINNER;
    int bl = i / DINNER;          // b*4096 + l
    int l  = bl & 4095;
    float acc = cb[c];
#pragma unroll
    for (int kk = 0; kk < 4; ++kk) {
      int ls = l + kk - 3;
      if (ls >= 0) acc += xz[(size_t)(bl + kk - 3) * NXZ + c] * cw[c*4 + kk];
    }
    xc[i] = (_Float16)siluf(acc);
  }
}

// ---------------------------------------------------------------- silu(z) gate cache
__global__ void k_zsilu(const float* __restrict__ xz, _Float16* __restrict__ zs, int total) {
  int stride = gridDim.x * blockDim.x;
  for (int i = blockIdx.x * blockDim.x + threadIdx.x; i < total; i += stride) {
    int c  = i % DINNER;
    int bl = i / DINNER;
    zs[i] = (_Float16)siluf(xz[(size_t)bl * NXZ + DINNER + c]);
  }
}

__global__ void k_cvt16(const float* __restrict__ a, _Float16* __restrict__ o, int n) {
  int stride = gridDim.x * blockDim.x;
  for (int i = blockIdx.x * blockDim.x + threadIdx.x; i < n; i += stride)
    o[i] = (_Float16)a[i];
}

// ---------------------------------------------------------------- selective scan
// One thread per (b, d) channel; 16 states in registers; B_t/C_t broadcast via
// double-buffered LDS (one barrier per step). grid = (3, 8), block = 256.
__global__ __launch_bounds__(256) void k_scan(const float* __restrict__ dt,
    const float* __restrict__ xdb, const _Float16* __restrict__ xc,
    const _Float16* __restrict__ zs, const float* __restrict__ Alog,
    const float* __restrict__ Dp, _Float16* __restrict__ y) {
  __shared__ float sBC[2][32];
  const int b = blockIdx.y;
  const int d = (blockIdx.x << 8) + threadIdx.x;
  float A[DSTATE], h[DSTATE];
#pragma unroll
  for (int s = 0; s < DSTATE; ++s) { A[s] = -__expf(Alog[d*DSTATE + s]); h[s] = 0.f; }
  const float dpar = Dp[d];
  if (threadIdx.x < 32)
    sBC[0][threadIdx.x] = xdb[(size_t)(b * LSEQ) * NXDB + 24 + threadIdx.x];
  __syncthreads();
  for (int t = 0; t < LSEQ; ++t) {
    if (threadIdx.x < 32 && t + 1 < LSEQ)
      sBC[(t + 1) & 1][threadIdx.x] = xdb[(size_t)(b * LSEQ + t + 1) * NXDB + 24 + threadIdx.x];
    const float* cur = sBC[t & 1];
    size_t idx = (size_t)(b * LSEQ + t) * DINNER + d;
    float dtv = dt[idx];
    float xv  = (float)xc[idx];
    float dx  = dtv * xv;
    float yv  = 0.f;
#pragma unroll
    for (int s = 0; s < DSTATE; ++s) {
      h[s] = h[s] * __expf(dtv * A[s]) + dx * cur[s];
      yv  += h[s] * cur[16 + s];
    }
    yv = (yv + xv * dpar) * (float)zs[idx];
    y[idx] = (_Float16)yv;
    __syncthreads();
  }
}

// ---------------------------------------------------------------- 8-way gather-sum
__global__ __launch_bounds__(128) void k_gather(const float* __restrict__ hm,
    const int* __restrict__ g, float* __restrict__ ho) {
  int r = blockIdx.x;              // b*4096 + l
  int b = r >> 12;
  int gi[8];
#pragma unroll
  for (int i = 0; i < 8; ++i) gi[i] = g[r*8 + i];
  for (int d = threadIdx.x; d < DMODEL; d += 128) {
    float s = 0.f;
#pragma unroll
    for (int i = 0; i < 8; ++i)
      s += hm[((size_t)(b * LSEQ + gi[i])) * DMODEL + d];
    ho[(size_t)r * DMODEL + d] = s;
  }
}

// ================================================================ host
extern "C" void kernel_launch(void* const* d_in, const int* in_sizes, int n_in,
                              void* d_out, int out_size, void* d_ws, size_t ws_size,
                              hipStream_t stream) {
  (void)in_sizes; (void)n_in; (void)out_size; (void)ws_size;
  const float* ids  = (const float*)d_in[0];
  const float* pos  = (const float*)d_in[1];
  const float* eig  = (const float*)d_in[2];
  const float* Win  = (const float*)d_in[3];
  const float* cw   = (const float*)d_in[4];
  const float* cb   = (const float*)d_in[5];
  const float* Wx   = (const float*)d_in[6];
  const float* Wdt  = (const float*)d_in[7];
  const float* bdt  = (const float*)d_in[8];
  const float* Alog = (const float*)d_in[9];
  const float* Dpar = (const float*)d_in[10];
  const float* Wout = (const float*)d_in[11];
  const float* lnw  = (const float*)d_in[12];
  const float* lnb  = (const float*)d_in[13];
  const float* nfw  = (const float*)d_in[14];
  const float* nfb  = (const float*)d_in[15];
  const int*   rev  = (const int*)d_in[17];

  char* ws = (char*)d_ws;
  size_t off = 0;
  auto alloc = [&](size_t bytes) -> char* {
    char* p = ws + off;
    off = (off + bytes + 255) & ~(size_t)255;
    return p;
  };
  float*    res    = (float*)   alloc(sizeof(float)    * MROWS * DMODEL);
  float*    hA     = (float*)   alloc(sizeof(float)    * MROWS * DMODEL);
  float*    hB     = (float*)   alloc(sizeof(float)    * MROWS * DMODEL);
  _Float16* hn16   = (_Float16*)alloc(sizeof(_Float16) * MROWS * DMODEL);
  float*    xz     = (float*)   alloc(sizeof(float)    * (size_t)MROWS * NXZ);
  _Float16* xc16   = (_Float16*)alloc(sizeof(_Float16) * (size_t)MROWS * DINNER);
  _Float16* zs16   = (_Float16*)alloc(sizeof(_Float16) * (size_t)MROWS * DINNER);
  float*    xdb    = (float*)   alloc(sizeof(float)    * (size_t)MROWS * NXDB);
  _Float16* xdb16  = (_Float16*)alloc(sizeof(_Float16) * (size_t)MROWS * NXDB);
  float*    dtb    = (float*)   alloc(sizeof(float)    * (size_t)MROWS * DINNER);
  _Float16* y16    = (_Float16*)alloc(sizeof(_Float16) * (size_t)MROWS * DINNER);
  _Float16* Win16  = (_Float16*)alloc(sizeof(_Float16) * 4 * 1536 * 384);
  _Float16* Wx16   = (_Float16*)alloc(sizeof(_Float16) * 4 * 64 * 768);
  _Float16* Wdt16  = (_Float16*)alloc(sizeof(_Float16) * 4 * 768 * 32);
  _Float16* Wout16 = (_Float16*)alloc(sizeof(_Float16) * 4 * 384 * 768);
  int*      perm   = (int*)     alloc(sizeof(int) * 32 * 512);
  int*      inv    = (int*)     alloc(sizeof(int) * 32 * 512);
  int*      gidx   = (int*)     alloc(sizeof(int) * (size_t)MROWS * 8);

  // one-time prep
  k_init<<<4096, 256, 0, stream>>>(ids, pos, hA, MROWS * DMODEL);
  k_cvtw<<<4096, 256, 0, stream>>>(Win, Wx, Wdt, Wout, Win16, Wx16, Wdt16, Wout16);
  k_sort<<<32, 512, 0, stream>>>(eig, perm, inv);
  k_gidx<<<128, 256, 0, stream>>>(perm, inv, rev, gidx);

  float* hm = xz;  // reuse xz as mamba-output buffer (free by out-proj time)

  for (int layer = 0; layer < 4; ++layer) {
    float* hin  = (layer & 1) ? hB : hA;
    float* hout = (layer & 1) ? hA : hB;

    k_ln<<<4096, 256, 0, stream>>>(hin, res, hn16,
                                   lnw + layer * DMODEL, lnb + layer * DMODEL, layer == 0);
    // in-proj: (32768 x 384) x (1536 x 384)^T
    k_gemm<0><<<dim3(MROWS/128, NXZ/16), 256, 0, stream>>>(
        hn16, Win16 + (size_t)layer * 1536 * 384, xz, MROWS, NXZ, 384, nullptr);
    k_conv<<<8192, 256, 0, stream>>>(xz, cw + layer * DINNER * 4, cb + layer * DINNER,
                                     xc16, MROWS * DINNER);
    k_zsilu<<<8192, 256, 0, stream>>>(xz, zs16, MROWS * DINNER);
    // x-proj: (32768 x 768) x (64 x 768)^T  (56 padded to 64)
    k_gemm<0><<<dim3(MROWS/128, NXDB/16), 256, 0, stream>>>(
        xc16, Wx16 + (size_t)layer * 64 * 768, xdb, MROWS, NXDB, 768, nullptr);
    k_cvt16<<<2048, 256, 0, stream>>>(xdb, xdb16, MROWS * NXDB);
    // dt-proj + softplus(.+b_dt): (32768 x 32) x (768 x 32)^T  (K 24 padded to 32)
    k_gemm<1><<<dim3(MROWS/128, DINNER/16), 256, 0, stream>>>(
        xdb16, Wdt16 + (size_t)layer * 768 * 32, dtb, MROWS, DINNER, KDT,
        bdt + layer * DINNER);
    // selective scan + D skip + z gate
    k_scan<<<dim3(3, 8), 256, 0, stream>>>(dtb, xdb, xc16, zs16,
                                           Alog + (size_t)layer * DINNER * DSTATE,
                                           Dpar + layer * DINNER, y16);
    // out-proj: (32768 x 768) x (384 x 768)^T
    k_gemm<0><<<dim3(MROWS/128, DMODEL/16), 256, 0, stream>>>(
        y16, Wout16 + (size_t)layer * 384 * 768, hm, MROWS, DMODEL, 768, nullptr);
    // cross-merge + eigen re-sort + reverse concat (fixed 8-way gather-sum)
    k_gather<<<MROWS, 128, 0, stream>>>(hm, gidx, hout);
  }

  // final: residual + LN with nf_w/nf_b  (layer 3 wrote into hA)
  k_ln_final<<<4096, 256, 0, stream>>>(hA, res, (float*)d_out, nfw, nfb);
}
